// HMMEstimator_4982162063823
// MI455X (gfx1250) — compile-verified, matrix operand order checked by reading, not run on hardware
//
#include <hip/hip_runtime.h>

typedef __attribute__((ext_vector_type(2)))  float    v2f;
typedef __attribute__((ext_vector_type(8)))  float    v8f;
typedef __attribute__((ext_vector_type(16))) _Float16 v16h;

#define HH   48
#define WW   48
#define NS   3
#define NOBS 16
#define HW   (HH * WW)          // 2304
#define JCOLS (HW * NS)         // 6912

// ---------------- Kernel A: zero-fill the whole output (u' + dense J) -------
// 191 MB of stores -> the bandwidth-bound bulk of the op (~8 us floor at
// 23.3 TB/s). 4x B128 stores per lane, coalesced per wave.
__global__ __launch_bounds__(256) void hmm_zero(float4* __restrict__ out4,
                                                float* __restrict__ outf,
                                                long long n4, long long n) {
  const float4 z4 = make_float4(0.f, 0.f, 0.f, 0.f);
  long long base = (long long)blockIdx.x * (blockDim.x * 4) + threadIdx.x;
  #pragma unroll
  for (int r = 0; r < 4; ++r) {
    long long i = base + (long long)r * blockDim.x;
    if (i < n4) out4[i] = z4;
  }
  if (base == 0) {                    // tail guard (out_size is in fact /4)
    for (long long k = n4 * 4; k < n; ++k) outf[k] = 0.f;
  }
}

// ---------------- Kernel B: observation update via WMMA ---------------------
// logits(2304x3) = obs(2304x16) @ W_lin^T(16x3); one wave per 16-row tile,
// four chained V_WMMA_F32_16X16X4_F32 (K = 4*4 = 16), B zero-padded to N=16.
// B loads use a clamped index + select so the WMMA loop is branch-free
// (EXEC stays all-1s, no saveexec round-trips between WMMAs).
__global__ __launch_bounds__(32) void hmm_obs_update(
    const float* __restrict__ obs, const float* __restrict__ u_k,
    const float* __restrict__ W_lin, const float* __restrict__ b_lin,
    float* __restrict__ Z) {
  __shared__ float sL[16 * NS];
  const int lane = threadIdx.x;       // wave32
  const int half = lane >> 4;         // 0: K lo pair / rows 0-7 of C
  const int l15  = lane & 15;
  const int rowBase = blockIdx.x << 4;
  const int nClamp  = (l15 < NS) ? l15 : 0;   // always-in-bounds W_lin row
  const float nMask = (l15 < NS) ? 1.0f : 0.0f;

  v8f acc = {};
#if __has_builtin(__builtin_amdgcn_wmma_f32_16x16x4_f32)
  // A 16x4 f32 layout: lanes 0-15 hold K=0,1 ; lanes 16-31 hold K=2,3 (M=l15)
  #pragma unroll
  for (int kc = 0; kc < 4; ++kc) {
    const int k0 = kc * 4 + half * 2;
    v2f a, b;
    a.x = obs[(rowBase + l15) * NOBS + k0];
    a.y = obs[(rowBase + l15) * NOBS + k0 + 1];
    // B[k][n] = (n < 3) ? W_lin[n][k] : 0   (N = l15); unconditional load +
    // v_cndmask-style select keeps the loop branch-free.
    b.x = nMask * W_lin[nClamp * NOBS + k0];
    b.y = nMask * W_lin[nClamp * NOBS + k0 + 1];
    acc = __builtin_amdgcn_wmma_f32_16x16x4_f32(false, a, false, b,
                                                (short)0, acc, false, false);
  }
#else
  // Fallback: single f16 WMMA with K padded 16->32 (zeros).
  v16h a = {}, b = {};
  #pragma unroll
  for (int k = 0; k < 8; ++k)         // lanes 0-15: K 0-7, lanes 16-31: K 8-15
    a[k] = (_Float16)obs[(rowBase + l15) * NOBS + half * 8 + k];
  const float bMask = (half == 0) ? nMask : 0.0f;
  #pragma unroll
  for (int k = 0; k < 16; ++k)        // B: lanes 0-15 hold K=0..15, col = l15
    b[k] = (_Float16)(bMask * W_lin[nClamp * NOBS + k]);
  acc = __builtin_amdgcn_wmma_f32_16x16x32_f16(false, a, false, b,
                                               (short)0, acc, false, false);
#endif

  // C/D layout: acc[r] = logits[rowBase + r + half*8][n = l15]
  if (l15 < NS) {
    const float bl = b_lin[l15];
    #pragma unroll
    for (int r = 0; r < 8; ++r)
      sL[(r + half * 8) * NS + l15] = acc[r] + bl;
  }
  __syncthreads();

  if (lane < 16) {
    const int row = rowBase + lane;
    float x0 = sL[lane * 3 + 0], x1 = sL[lane * 3 + 1], x2 = sL[lane * 3 + 2];
    float mx = fmaxf(x0, fmaxf(x1, x2));
    float e0 = __expf(x0 - mx), e1 = __expf(x1 - mx), e2 = __expf(x2 - mx);
    float inv = 1.0f / (e0 + e1 + e2);
    float Bu0 = (e0 * inv) * u_k[row * 3 + 0];
    float Bu1 = (e1 * inv) * u_k[row * 3 + 1];
    float Bu2 = (e2 * inv) * u_k[row * 3 + 2];
    float ibu = 1.0f / (Bu0 + Bu1 + Bu2);
    Z[row * 3 + 0] = Bu0 * ibu;
    Z[row * 3 + 1] = Bu1 * ibu;
    Z[row * 3 + 2] = Bu2 * ibu;
  }
}

// ---------------- Kernel C: conv3x3 + softmax + Jacobian scatter ------------
__global__ __launch_bounds__(256) void hmm_transition_jac(
    const float* __restrict__ Z, const float* __restrict__ W_conv,
    const float* __restrict__ b_conv, float* __restrict__ out) {
  __shared__ float sW[NS * NS * 9];   // OIHW: [m][c][ki][kj]
  __shared__ float sB[NS];
  const int tid = threadIdx.x;
  if (tid < NS * NS * 9) sW[tid] = W_conv[tid];
  if (tid < NS) sB[tid] = b_conv[tid];
  __syncthreads();

  const int pix = blockIdx.x * blockDim.x + tid;
  if (pix >= HW) return;
  const int i = pix / WW, j = pix % WW;

  // conv2d(3->3, k=3, pad=1), NCHW/OIHW
  float y0 = sB[0], y1 = sB[1], y2 = sB[2];
  #pragma unroll
  for (int ki = 0; ki < 3; ++ki) {
    const int ii = i + ki - 1;
    if (ii < 0 || ii >= HH) continue;
    #pragma unroll
    for (int kj = 0; kj < 3; ++kj) {
      const int jj = j + kj - 1;
      if (jj < 0 || jj >= WW) continue;
      const int zb = (ii * WW + jj) * NS;
      #pragma unroll
      for (int c = 0; c < NS; ++c) {
        const float z = Z[zb + c];
        y0 += sW[(0 * NS + c) * 9 + ki * 3 + kj] * z;
        y1 += sW[(1 * NS + c) * 9 + ki * 3 + kj] * z;
        y2 += sW[(2 * NS + c) * 9 + ki * 3 + kj] * z;
      }
    }
  }
  // channel softmax -> u_{k+1}
  float mx = fmaxf(y0, fmaxf(y1, y2));
  float e0 = __expf(y0 - mx), e1 = __expf(y1 - mx), e2 = __expf(y2 - mx);
  float inv = 1.0f / (e0 + e1 + e2);
  float u[3] = { e0 * inv, e1 * inv, e2 * inv };
  out[pix * 3 + 0] = u[0];
  out[pix * 3 + 1] = u[1];
  out[pix * 3 + 2] = u[2];

  // softmax Jacobian S[c][m] = u_c (delta_cm - u_m)
  float S[3][3];
  #pragma unroll
  for (int c = 0; c < 3; ++c)
    #pragma unroll
    for (int m = 0; m < 3; ++m)
      S[c][m] = u[c] * (((c == m) ? 1.0f : 0.0f) - u[m]);

  // dPhi[i,j,c]/dZ[i+di, j+dj, c'] = sum_m S[c][m] * W[m][c'][di+1][dj+1]
  #pragma unroll
  for (int di = -1; di <= 1; ++di) {
    const int ii = i + di;
    if (ii < 0 || ii >= HH) continue;
    #pragma unroll
    for (int dj = -1; dj <= 1; ++dj) {
      const int jj = j + dj;
      if (jj < 0 || jj >= WW) continue;
      const int colb = (ii * WW + jj) * NS;
      const int kk = (di + 1) * 3 + (dj + 1);
      #pragma unroll
      for (int c = 0; c < 3; ++c) {
        float* Jr = out + (size_t)JCOLS * (size_t)(1 + pix * 3 + c) + colb;
        #pragma unroll
        for (int cp = 0; cp < 3; ++cp) {
          Jr[cp] = S[c][0] * sW[(0 * NS + cp) * 9 + kk]
                 + S[c][1] * sW[(1 * NS + cp) * 9 + kk]
                 + S[c][2] * sW[(2 * NS + cp) * 9 + kk];
        }
      }
    }
  }
}

extern "C" void kernel_launch(void* const* d_in, const int* in_sizes, int n_in,
                              void* d_out, int out_size, void* d_ws, size_t ws_size,
                              hipStream_t stream) {
  const float* obs    = (const float*)d_in[0];  // (48,48,16)
  const float* u_k    = (const float*)d_in[1];  // (48,48,3)
  const float* W_lin  = (const float*)d_in[2];  // (3,16)
  const float* b_lin  = (const float*)d_in[3];  // (3,)
  const float* W_conv = (const float*)d_in[4];  // (3,3,3,3) OIHW
  const float* b_conv = (const float*)d_in[5];  // (3,)
  float* out = (float*)d_out;                   // [u' (6912) | J (6912*6912)]
  float* Z   = (float*)d_ws;                    // 2304*3 floats scratch

  const long long n  = (long long)out_size;     // 47,782,656 floats
  const long long n4 = n / 4;
  const int tb = 256;
  const int gz = (int)((n4 + (long long)tb * 4 - 1) / ((long long)tb * 4));

  hmm_zero<<<gz, tb, 0, stream>>>((float4*)d_out, out, n4, n);
  hmm_obs_update<<<HW / 16, 32, 0, stream>>>(obs, u_k, W_lin, b_lin, Z);
  hmm_transition_jac<<<(HW + tb - 1) / tb, tb, 0, stream>>>(Z, W_conv, b_conv, out);
}
